// SpatialAttention_13434657702207
// MI455X (gfx1250) — matrix operand order repaired and checked by reading, not compiled, not run on hardware
//
#include <hip/hip_runtime.h>

typedef __attribute__((ext_vector_type(16))) _Float16 v16h;
typedef __attribute__((ext_vector_type(8)))  _Float16 v8h;
typedef __attribute__((ext_vector_type(8)))  float    v8f;

constexpr int B = 4, H = 64, W = 64, C = 128, P = H * W;  // P = 4096

// ---------------------------------------------------------------------------
// K0: channel mean/max pool  input[B,H,W,C] -> xsp[B,P,2]
// ---------------------------------------------------------------------------
__global__ void k_pool(const float* __restrict__ in, float* __restrict__ xsp) {
  int idx = blockIdx.x * blockDim.x + threadIdx.x;  // b*P + p
  if (idx >= B * P) return;
  const float4* ip = (const float4*)(in + (size_t)idx * C);
  float s = 0.f, m = -3.4e38f;
#pragma unroll 8
  for (int q = 0; q < C / 4; q++) {
    float4 v = ip[q];
    s += v.x + v.y + v.z + v.w;
    m = fmaxf(m, fmaxf(fmaxf(v.x, v.y), fmaxf(v.z, v.w)));
  }
  xsp[idx * 2 + 0] = s * (1.0f / C);
  xsp[idx * 2 + 1] = m;
}

// ---------------------------------------------------------------------------
// K1: build 3x3x2 patches (d = (kh*3+kw)*2+c, padded 18->32) and scatter them
// into WMMA register layouts:
//   QA: A-layout of xf          [B][P/16 tiles][32 lanes][16 halves]
//   KB: B-layout of l2-norm xf  [B][P/16 tiles][32 lanes][16 halves]
//   VB: B-layout of xf for PV   [B][P/32 iters][2 ntiles][32 lanes][16 halves]
//       with a ones-column at c=18 so the PV WMMA also accumulates row sums.
// Also: mm[p] = all-9-valid mask, norms[b*P+p] = ||xf|| (static softmax shift,
// provably >= every logit of row p since keys are unit-norm and mask logits=0).
// ---------------------------------------------------------------------------
__global__ void k_patches(const float* __restrict__ xsp, const float* __restrict__ mask,
                          float* __restrict__ mmv, float* __restrict__ norms,
                          _Float16* __restrict__ QA, _Float16* __restrict__ KB,
                          _Float16* __restrict__ VB) {
  int idx = blockIdx.x * blockDim.x + threadIdx.x;
  if (idx >= B * P) return;
  int b = idx / P, p = idx % P, i = p / W, j = p % W;
  float v[32];
#pragma unroll
  for (int k = 0; k < 32; k++) v[k] = 0.f;
#pragma unroll
  for (int kh = 0; kh < 3; kh++)
#pragma unroll
    for (int kw = 0; kw < 3; kw++) {
      int ii = i + kh - 1, jj = j + kw - 1;
      if (ii >= 0 && ii < H && jj >= 0 && jj < W) {
        const float* s = xsp + ((size_t)(b * P + ii * W + jj)) * 2;
        v[(kh * 3 + kw) * 2 + 0] = s[0];
        v[(kh * 3 + kw) * 2 + 1] = s[1];
      }
    }
  float nrm = 0.f;
#pragma unroll
  for (int k = 0; k < 18; k++) nrm += v[k] * v[k];
  float nn = sqrtf(nrm);
  float inv = 1.0f / fmaxf(nn, 1e-4f);
  norms[idx] = nn;

  int tile = p >> 4, n = p & 15;
  size_t qbase = ((size_t)(b * (P / 16) + tile)) * 32;
  // A layout (16x32 f16): lane<16 holds K in {0..7,16..23}, lane>=16 K in {8..15,24..31}
#pragma unroll
  for (int k = 0; k < 32; k++) {
    int lane = n + ((k & 8) ? 16 : 0);
    int pos = ((k & 16) ? 8 : 0) + (k & 7);
    QA[(qbase + lane) * 16 + pos] = (_Float16)v[k];
  }
  // B layout (32x16 f16): lane<16 holds K in {0..7,16..23}, lane>=16 K in {8..15,24..31}; lane%16 = N
#pragma unroll
  for (int k = 0; k < 32; k++) {
    int lane = n + ((k & 8) ? 16 : 0);
    int pos = ((k & 16) ? 8 : 0) + (k & 7);
    KB[(qbase + lane) * 16 + pos] = (_Float16)(k < 18 ? v[k] * inv : 0.f);
  }
  int it = p >> 5, pl = p & 31;
#pragma unroll
  for (int c = 0; c < 32; c++) {
    int lane = (c & 15) + ((pl & 8) ? 16 : 0);
    int pos = ((pl & 16) ? 8 : 0) + (pl & 7);
    int ntile = c >> 4;
    float vv = (c < 18) ? v[c] : ((c == 18) ? 1.0f : 0.0f);  // ones column -> row sums
    VB[((((size_t)(b * (P / 32) + it)) * 2 + ntile) * 32 + lane) * 16 + pos] = (_Float16)vv;
  }
  if (b == 0) {
    float ok = 1.f;
    for (int kh = 0; kh < 3; kh++)
      for (int kw = 0; kw < 3; kw++) {
        int ii = i + kh - 1, jj = j + kw - 1;
        float mv = (ii >= 0 && ii < H && jj >= 0 && jj < W) ? mask[ii * W + jj] : 0.f;
        ok = fminf(ok, (mv == 1.0f) ? 1.f : 0.f);
      }
    mmv[p] = ok;
  }
}

// ---------------------------------------------------------------------------
// K2: fused flash attention with a STATIC per-row softmax shift M[n]=||xf_n||.
// No per-iteration cross-lane reductions: the ones-column of V makes the PV
// WMMA accumulate the masked numerator sums, and the denominator's masked-zero
// terms are Nmask*exp(-M[n]) with Nmask accumulated scalar-wise. Inner loop:
// 4 WMMAs + 16 exps + 18 DS ops, nothing else.
// ---------------------------------------------------------------------------
__global__ __launch_bounds__(256) void k_flash(const _Float16* __restrict__ QA,
                                               const _Float16* __restrict__ KB,
                                               const _Float16* __restrict__ VB,
                                               const float* __restrict__ mmv,
                                               const float* __restrict__ norms,
                                               float* __restrict__ otmp) {
  __shared__ _Float16 lds[8][16 * 32];
  int lane = threadIdx.x & 31;
  int wv = threadIdx.x >> 5;
  int rt = blockIdx.x * 8 + wv;  // global 16-row tile id
  int b = rt >> 8;               // / (P/16)
  int tile = rt & 255;
  int n0 = tile * 16;
  const v16h* QAp = (const v16h*)QA;
  const v16h* KBp = (const v16h*)KB;
  const v16h* VBp = (const v16h*)VB;
  int colh = lane & 15;
  int rowoff = (lane < 16) ? 0 : 8;

  v16h qa = QAp[((size_t)(b * (P / 16) + tile)) * 32 + lane];
  float M[8];
#pragma unroll
  for (int r = 0; r < 8; r++) M[r] = norms[b * P + n0 + r + rowoff];

  v8f o0 = {0.f, 0.f, 0.f, 0.f, 0.f, 0.f, 0.f, 0.f};
  v8f o1 = {0.f, 0.f, 0.f, 0.f, 0.f, 0.f, 0.f, 0.f};
  float cnt = 0.f;

  for (int it = 0; it < P / 32; ++it) {
    v16h kb0 = KBp[((size_t)(b * (P / 16) + it * 2)) * 32 + lane];
    v16h kb1 = KBp[((size_t)(b * (P / 16) + it * 2 + 1)) * 32 + lane];
    v8f zc = {0.f, 0.f, 0.f, 0.f, 0.f, 0.f, 0.f, 0.f};
    v8f s0 = __builtin_amdgcn_wmma_f32_16x16x32_f16(false, qa, false, kb0, (short)0, zc, false, false);
    v8f s1 = __builtin_amdgcn_wmma_f32_16x16x32_f16(false, qa, false, kb1, (short)0, zc, false, false);
    float mm0 = mmv[it * 32 + colh];
    float mm1 = mmv[it * 32 + 16 + colh];
    cnt += mm0 + mm1;
#pragma unroll
    for (int r = 0; r < 8; r++) {
      // z = score*mm (masked cols -> z=0, exact ref semantics); numerator re-masked
      float p0 = __expf(__builtin_fmaf(s0[r], mm0, -M[r])) * mm0;
      float p1 = __expf(__builtin_fmaf(s1[r], mm1, -M[r])) * mm1;
      lds[wv][(r + rowoff) * 32 + colh] = (_Float16)p0;
      lds[wv][(r + rowoff) * 32 + colh + 16] = (_Float16)p1;
    }
    // C-layout -> A-layout via LDS: lane<16 reads cols {0..7,16..23} of its row
    int arow = lane & 15;
    int abase = (lane < 16) ? 0 : 8;
    v8h lo = *(const v8h*)&lds[wv][arow * 32 + abase];
    v8h hi = *(const v8h*)&lds[wv][arow * 32 + abase + 16];
    v16h apr = __builtin_shufflevector(lo, hi, 0, 1, 2, 3, 4, 5, 6, 7, 8, 9, 10, 11, 12, 13, 14, 15);
    v16h vb0 = VBp[(((size_t)(b * (P / 32) + it)) * 2 + 0) * 32 + lane];
    v16h vb1 = VBp[(((size_t)(b * (P / 32) + it)) * 2 + 1) * 32 + lane];
    o0 = __builtin_amdgcn_wmma_f32_16x16x32_f16(false, apr, false, vb0, (short)0, o0, false, false);
    o1 = __builtin_amdgcn_wmma_f32_16x16x32_f16(false, apr, false, vb1, (short)0, o1, false, false);
  }

  // total valid-column count (lanes 0..15 tile the column space exactly once)
  cnt += __shfl_xor(cnt, 1, 16);
  cnt += __shfl_xor(cnt, 2, 16);
  cnt += __shfl_xor(cnt, 4, 16);
  cnt += __shfl_xor(cnt, 8, 16);
  float nmask = (float)P - cnt;

#pragma unroll
  for (int r = 0; r < 8; r++) {
    // row sum lives in accumulator column c=18 (tile1, N=2) -> broadcast from lane 2
    float sum = __shfl(o1[r], 2, 16);
    float D = sum + nmask * __expf(-M[r]);
    float invD = 1.0f / D;
    int n = n0 + r + rowoff;
    otmp[((size_t)(b * P + n)) * 18 + colh] = o0[r] * invD;
    if (colh < 2) otmp[((size_t)(b * P + n)) * 18 + 16 + colh] = o1[r] * invD;
  }
}

// ---------------------------------------------------------------------------
// K3a: transpose-conv gather (y channel 0 only) + 7x7 gate conv + sigmoid
// coef[b,p] = {sigmoid(g), y0}
// ---------------------------------------------------------------------------
__global__ void k_spatial(const float* __restrict__ otmp, const float* __restrict__ xsp,
                          const float* __restrict__ cw, float* __restrict__ coef) {
  int idx = blockIdx.x * blockDim.x + threadIdx.x;
  if (idx >= B * P) return;
  int b = idx / P, p = idx % P, i = p / W, j = p % W;
  float y0 = 0.f;
#pragma unroll
  for (int kh = 0; kh < 3; kh++)
#pragma unroll
    for (int kw = 0; kw < 3; kw++) {
      int ii = i + 1 - kh, jj = j + 1 - kw;
      if (ii >= 0 && ii < H && jj >= 0 && jj < W)
        y0 += otmp[((size_t)(b * P + ii * W + jj)) * 18 + (kh * 3 + kw) * 2];
    }
  y0 *= 0.25f;
  float g = 0.f;
  for (int kh = 0; kh < 7; kh++)
    for (int kw = 0; kw < 7; kw++) {
      int ii = i + kh - 3, jj = j + kw - 3;
      if (ii >= 0 && ii < H && jj >= 0 && jj < W) {
        const float* s = xsp + ((size_t)(b * P + ii * W + jj)) * 2;
        g += s[0] * cw[(kh * 7 + kw) * 2 + 0] + s[1] * cw[(kh * 7 + kw) * 2 + 1];
      }
    }
  float sig = 1.0f / (1.0f + __expf(-g));
  coef[idx * 2 + 0] = sig;
  coef[idx * 2 + 1] = y0;
}

// ---------------------------------------------------------------------------
// K3b: out = sigmoid(g) * (input + (y0 + bias) * (1 - mask)), float4 streams
// ---------------------------------------------------------------------------
__global__ void k_final(const float* __restrict__ in, const float* __restrict__ mask,
                        const float* __restrict__ bias, const float* __restrict__ coef,
                        float* __restrict__ out) {
  int t = blockIdx.x * blockDim.x + threadIdx.x;
  int nq = B * P * C / 4;
  if (t >= nq) return;
  int qc = C / 4;
  int pix = t / qc;  // b*P + p
  int cq = (t % qc) * 4;
  int p = pix % P;
  float sig = coef[pix * 2 + 0], y0 = coef[pix * 2 + 1];
  float im = 1.0f - mask[p];
  float4 i4 = *(const float4*)(in + (size_t)pix * C + cq);
  float4 b4 = *(const float4*)(bias + (size_t)p * C + cq);
  float4 o;
  o.x = sig * (i4.x + (y0 + b4.x) * im);
  o.y = sig * (i4.y + (y0 + b4.y) * im);
  o.z = sig * (i4.z + (y0 + b4.z) * im);
  o.w = sig * (i4.w + (y0 + b4.w) * im);
  *(float4*)(out + (size_t)pix * C + cq) = o;
}

extern "C" void kernel_launch(void* const* d_in, const int* in_sizes, int n_in,
                              void* d_out, int out_size, void* d_ws, size_t ws_size,
                              hipStream_t stream) {
  const float* input = (const float*)d_in[0];
  const float* mask = (const float*)d_in[1];
  const float* bias = (const float*)d_in[2];
  const float* convw = (const float*)d_in[3];
  float* out = (float*)d_out;

  float* xsp = (float*)d_ws;                      // B*P*2
  float* mmv = xsp + B * P * 2;                   // P
  float* otmp = mmv + P;                          // B*P*18
  float* coef = otmp + (size_t)B * P * 18;        // B*P*2
  float* norms = coef + B * P * 2;                // B*P
  _Float16* QA = (_Float16*)(norms + B * P);      // B*P*32 halves
  _Float16* KB = QA + (size_t)B * P * 32;
  _Float16* VB = KB + (size_t)B * P * 32;

  k_pool<<<(B * P) / 256, 256, 0, stream>>>(input, xsp);
  k_patches<<<(B * P) / 256, 256, 0, stream>>>(xsp, mask, mmv, norms, QA, KB, VB);
  k_flash<<<(B * P / 16) / 8, 256, 0, stream>>>(QA, KB, VB, mmv, norms, otmp);
  k_spatial<<<(B * P) / 256, 256, 0, stream>>>(otmp, xsp, convw, coef);
  k_final<<<(B * P * C / 4) / 256, 256, 0, stream>>>(input, mask, bias, coef, out);
}
